// SingleHeadAttention_27685359190643
// MI455X (gfx1250) — matrix operand order repaired and checked by reading, not compiled, not run on hardware
//
#include <hip/hip_runtime.h>

// ---------------------------------------------------------------------------
// Types
// ---------------------------------------------------------------------------
typedef __bf16 bf16_t;
typedef __attribute__((ext_vector_type(16))) __bf16        v16bf;
typedef __attribute__((ext_vector_type(8)))  float         v8f;
typedef __attribute__((ext_vector_type(8)))  unsigned int  v8u;

static __device__ __forceinline__ v8f zero8() {
    v8f z = {0.f, 0.f, 0.f, 0.f, 0.f, 0.f, 0.f, 0.f};
    return z;
}

// Load a 16-element bf16 fragment as two 16-byte chunks (global or LDS).
static __device__ __forceinline__ v16bf load_frag(const bf16_t* p0, const bf16_t* p1) {
    const uint4 a = *reinterpret_cast<const uint4*>(p0);
    const uint4 b = *reinterpret_cast<const uint4*>(p1);
    v8u t = {a.x, a.y, a.z, a.w, b.x, b.y, b.z, b.w};
    return __builtin_bit_cast(v16bf, t);
}

// Async global -> LDS 16-byte copy (CDNA5, tracked by ASYNCcnt).
// LDS target address = low 32 bits of the generic shared pointer
// (flat->LDS translation truncates to addr[31:0] = workgroup-relative offset).
static __device__ __forceinline__ void async_copy_b128(const bf16_t* gsrc, bf16_t* ldst) {
    const unsigned l = (unsigned)(unsigned long long)(const void*)ldst;
    asm volatile("global_load_async_to_lds_b128 %0, %1, off"
                 :: "v"(l), "v"(gsrc)
                 : "memory");
}

static __device__ __forceinline__ void wait_async_le5() {
    asm volatile("s_wait_asynccnt 0x5" ::: "memory");
}
static __device__ __forceinline__ void wait_async_le0() {
    asm volatile("s_wait_asynccnt 0x0" ::: "memory");
}

// ---------------------------------------------------------------------------
// Elementwise f32 -> bf16
// ---------------------------------------------------------------------------
__global__ void f32_to_bf16_kernel(const float* __restrict__ in,
                                   bf16_t* __restrict__ out, long n) {
    long i = (long)blockIdx.x * blockDim.x + threadIdx.x;
    const long stride = (long)gridDim.x * blockDim.x;
    for (; i < n; i += stride) out[i] = (bf16_t)in[i];
}

// ---------------------------------------------------------------------------
// Tiled transpose (+ convert to bf16).  in: [rows][cols] -> out: [cols][rows]
// rows, cols multiples of 32.  block (32,8), grid (cols/32, rows/32, batch)
// ---------------------------------------------------------------------------
template <typename TIN>
__global__ void transpose_to_bf16_kernel(const TIN* __restrict__ in,
                                         bf16_t* __restrict__ out,
                                         int rows, int cols,
                                         long inBatchStride, long outBatchStride) {
    __shared__ bf16_t tile[32][33];
    const long z = blockIdx.z;
    const TIN*   ip = in  + z * inBatchStride;
    bf16_t*      op = out + z * outBatchStride;
    const int c0 = blockIdx.x * 32;
    const int r0 = blockIdx.y * 32;
    const int tx = threadIdx.x, ty = threadIdx.y;
#pragma unroll
    for (int j = 0; j < 32; j += 8) {
        tile[ty + j][tx] = (bf16_t)(float)ip[(long)(r0 + ty + j) * cols + (c0 + tx)];
    }
    __syncthreads();
#pragma unroll
    for (int j = 0; j < 32; j += 8) {
        op[(long)(c0 + ty + j) * rows + (r0 + tx)] = tile[tx][ty + j];
    }
}

// ---------------------------------------------------------------------------
// bf16 WMMA GEMM with async global->LDS double buffering.
//   C[M][N] = scale * (A[M][K] @ BT[N][K]^T) + bias
// Block: 256 threads (8 waves). Block tile 64(M) x 256(N). Wave tile 32x64.
// LDS: two 20KB stages (A 64x32 + B 256x32, bf16).
// Per thread per stage: 5 async b128 copies (1 A chunk + 4 B chunks).
// grid = (N/256, M/64, batch);  K must be a multiple of 32 and >= 64.
// ---------------------------------------------------------------------------
__global__ __launch_bounds__(256, 2)
void gemm_bf16_wmma(const bf16_t* __restrict__ A,  long aBatch, int lda,
                    const bf16_t* __restrict__ BT, long bBatch, int ldb,
                    void* __restrict__ Cv,         long cBatch, int ldc,
                    const float* __restrict__ bias, float scale, int K,
                    int outIsBf16) {
    __shared__ bf16_t sA[2][64 * 32];
    __shared__ bf16_t sB[2][256 * 32];

    const int t    = threadIdx.x;
    const int lane = t & 31;
    const int wid  = t >> 5;
    const int half = lane >> 4;     // 0 or 1
    const int l16  = lane & 15;
    const long z   = blockIdx.z;

    const bf16_t* Ab = A  + z * aBatch;
    const bf16_t* Bb = BT + z * bBatch;

    const long mBlk = (long)blockIdx.y * 64;
    const long nBlk = (long)blockIdx.x * 256;
    const int  wm   = (wid >> 2) * 32;   // wave M origin within block tile
    const int  wn   = (wid & 3)  * 64;   // wave N origin within block tile

    // Per-thread copy assignment (constant across K-steps).
    const int aRow = t >> 2,        aCol = (t & 3) * 8;           // A: 1 chunk
    const bf16_t* aSrcRow = Ab + (mBlk + aRow) * (long)lda + aCol;

    v8f acc[2][4];
#pragma unroll
    for (int tm = 0; tm < 2; ++tm)
#pragma unroll
        for (int tn = 0; tn < 4; ++tn) acc[tm][tn] = zero8();

    const int nsteps = K >> 5;

    // ---- stage issue: copy block tile for K-offset k0 into buffers --------
    auto issue_stage = [&](int k0, int st) {
        async_copy_b128(aSrcRow + k0, &sA[st][aRow * 32 + aCol]);
#pragma unroll
        for (int j = 0; j < 4; ++j) {
            const int chunk = t + j * 256;
            const int n     = chunk >> 2;
            const int c     = (chunk & 3) * 8;
            async_copy_b128(Bb + (nBlk + n) * (long)ldb + k0 + c,
                            &sB[st][n * 32 + c]);
        }
    };

    issue_stage(0, 0);
    issue_stage(32, 1);

    for (int i = 0; i < nsteps; ++i) {
        if (i + 1 < nsteps) wait_async_le5();   // oldest stage (i) complete
        else                wait_async_le0();   // nothing newer in flight
        __syncthreads();                        // all waves' data visible

        const int st = i & 1;
        v16bf af[2], bfr[4];
#pragma unroll
        for (int tm = 0; tm < 2; ++tm) {
            const bf16_t* p = &sA[st][(wm + tm * 16 + l16) * 32 + 8 * half];
            af[tm] = load_frag(p, p + 16);
        }
#pragma unroll
        for (int tn = 0; tn < 4; ++tn) {
            const bf16_t* p = &sB[st][(wn + tn * 16 + l16) * 32 + 8 * half];
            bfr[tn] = load_frag(p, p + 16);
        }
#pragma unroll
        for (int tm = 0; tm < 2; ++tm)
#pragma unroll
            for (int tn = 0; tn < 4; ++tn)
                acc[tm][tn] = __builtin_amdgcn_wmma_f32_16x16x32_bf16(
                    false, af[tm], false, bfr[tn], (short)0, acc[tm][tn],
                    false, false);

        __syncthreads();                        // everyone done reading st
        if (i + 2 < nsteps) issue_stage((i + 2) * 32, st);
    }

    // Epilogue.  C fragment: VGPR r -> row m = 16*tm + 8*half + r, col = l16.
#pragma unroll
    for (int tm = 0; tm < 2; ++tm) {
#pragma unroll
        for (int tn = 0; tn < 4; ++tn) {
            const long col = nBlk + wn + tn * 16 + l16;
            const float bv = bias ? bias[col] : 0.f;
            const long mBase = mBlk + wm + tm * 16 + 8 * half;
            if (outIsBf16) {
                bf16_t* C = (bf16_t*)Cv + z * cBatch;
#pragma unroll
                for (int r = 0; r < 8; ++r)
                    C[(mBase + r) * (long)ldc + col] =
                        (bf16_t)(acc[tm][tn][r] * scale + bv);
            } else {
                float* C = (float*)Cv + z * cBatch;
#pragma unroll
                for (int r = 0; r < 8; ++r)
                    C[(mBase + r) * (long)ldc + col] =
                        acc[tm][tn][r] * scale + bv;
            }
        }
    }
}

// ---------------------------------------------------------------------------
// Row softmax over 2048 f32, writing bf16 probabilities in-place over the row.
// One 256-thread block per row.  All global reads happen before the first
// barrier; all global writes after the last -> in-place is safe.
// ---------------------------------------------------------------------------
__global__ __launch_bounds__(256)
void softmax_rows_inplace(float* __restrict__ S, int rowLen /* == 2048 */) {
    const long row = blockIdx.x;
    float* rp = S + row * (long)rowLen;
    const int t = threadIdx.x;

    float vals[8];
    float mx = -3.4e38f;
#pragma unroll
    for (int j = 0; j < 8; ++j) {
        vals[j] = rp[t + j * 256];
        mx = fmaxf(mx, vals[j]);
    }
    __shared__ float red[256];
    red[t] = mx;
    __syncthreads();
    for (int s = 128; s > 0; s >>= 1) {
        if (t < s) red[t] = fmaxf(red[t], red[t + s]);
        __syncthreads();
    }
    mx = red[0];
    __syncthreads();

    float sum = 0.f;
#pragma unroll
    for (int j = 0; j < 8; ++j) {
        vals[j] = __expf(vals[j] - mx);
        sum += vals[j];
    }
    red[t] = sum;
    __syncthreads();
    for (int s = 128; s > 0; s >>= 1) {
        if (t < s) red[t] += red[t + s];
        __syncthreads();
    }
    const float inv = 1.0f / red[0];

    bf16_t* P = (bf16_t*)rp;   // bf16 row occupies first half of the f32 row
#pragma unroll
    for (int j = 0; j < 8; ++j)
        P[t + j * 256] = (bf16_t)(vals[j] * inv);
}

// ---------------------------------------------------------------------------
// Host-side launch
// ---------------------------------------------------------------------------
extern "C" void kernel_launch(void* const* d_in, const int* in_sizes, int n_in,
                              void* d_out, int out_size, void* d_ws, size_t ws_size,
                              hipStream_t stream) {
    (void)in_sizes; (void)n_in; (void)out_size; (void)ws_size;

    const int B = 4, S = 2048, D = 1024;
    const long M   = (long)B * S;       // 8192
    const long XEL = M * D;             // 8,388,608

    const float* x  = (const float*)d_in[0];
    const float* Wq = (const float*)d_in[1];
    const float* bq = (const float*)d_in[2];
    const float* Wk = (const float*)d_in[3];
    const float* bk = (const float*)d_in[4];
    const float* Wv = (const float*)d_in[5];
    const float* bv = (const float*)d_in[6];
    const float* Wo = (const float*)d_in[7];
    const float* bo = (const float*)d_in[8];
    float* out = (float*)d_out;

    // ---- workspace layout (bytes) --------------------------------------
    char* ws = (char*)d_ws;
    const size_t SZ_X    = (size_t)XEL * 2;             // 16 MiB  bf16 x
    const size_t SZ_W    = (size_t)D * D * 2;           //  2 MiB  bf16 W^T
    const size_t SZ_QKV  = (size_t)XEL * 2;             // 16 MiB  each
    const size_t SZ_S    = (size_t)B * S * S * 4;       // 64 MiB  f32 scores

    size_t off = 0;
    bf16_t* xbf  = (bf16_t*)(ws + off); off += SZ_X;
    bf16_t* WqT  = (bf16_t*)(ws + off); off += SZ_W;
    bf16_t* WkT  = (bf16_t*)(ws + off); off += SZ_W;
    bf16_t* WvT  = (bf16_t*)(ws + off); off += SZ_W;
    bf16_t* WoT  = (bf16_t*)(ws + off); off += SZ_W;
    bf16_t* qb   = (bf16_t*)(ws + off); off += SZ_QKV;
    bf16_t* kb   = (bf16_t*)(ws + off); off += SZ_QKV;
    bf16_t* vb   = (bf16_t*)(ws + off); off += SZ_QKV;
    float*  Sb   = (float*)(ws + off);  off += SZ_S;
    // Safe stream-ordered reuse:
    bf16_t* vT   = xbf;   // x only read by QKV GEMMs, which precede the V transpose
    bf16_t* attn = qb;    // q only read by the scores GEMM, which precedes P@V^T

    // 1) x -> bf16
    f32_to_bf16_kernel<<<1024, 256, 0, stream>>>(x, xbf, XEL);

    // 2) W^T in bf16 (B-operand layout: row n holds contiguous K)
    {
        dim3 blk(32, 8), grd(D / 32, D / 32, 1);
        transpose_to_bf16_kernel<float><<<grd, blk, 0, stream>>>(Wq, WqT, D, D, 0, 0);
        transpose_to_bf16_kernel<float><<<grd, blk, 0, stream>>>(Wk, WkT, D, D, 0, 0);
        transpose_to_bf16_kernel<float><<<grd, blk, 0, stream>>>(Wv, WvT, D, D, 0, 0);
        transpose_to_bf16_kernel<float><<<grd, blk, 0, stream>>>(Wo, WoT, D, D, 0, 0);
    }

    // 3) Q/K/V projections: [8192x1024] = x @ W + b    (bf16 out)
    {
        dim3 grd(D / 256, M / 64, 1);
        gemm_bf16_wmma<<<grd, 256, 0, stream>>>(xbf, 0, D, WqT, 0, D,
                                                qb, 0, D, bq, 1.f, D, 1);
        gemm_bf16_wmma<<<grd, 256, 0, stream>>>(xbf, 0, D, WkT, 0, D,
                                                kb, 0, D, bk, 1.f, D, 1);
        gemm_bf16_wmma<<<grd, 256, 0, stream>>>(xbf, 0, D, WvT, 0, D,
                                                vb, 0, D, bv, 1.f, D, 1);
    }

    // 4) V^T per batch: [2048x1024] -> [1024x2048]   (into retired x region)
    {
        dim3 blk(32, 8), grd(D / 32, S / 32, B);
        transpose_to_bf16_kernel<bf16_t><<<grd, blk, 0, stream>>>(
            vb, vT, S, D, (long)S * D, (long)S * D);
    }

    // 5) scores = (q @ k^T) / 32   (f32 out; K already in B-operand layout)
    {
        dim3 grd(S / 256, S / 64, B);
        gemm_bf16_wmma<<<grd, 256, 0, stream>>>(qb, (long)S * D, D,
                                                kb, (long)S * D, D,
                                                Sb, (long)S * S, S,
                                                nullptr, 0.03125f, D, 0);
    }

    // 6) softmax rows, probabilities written bf16 in-place (lda becomes 2*S)
    softmax_rows_inplace<<<B * S, 256, 0, stream>>>(Sb, S);

    // 7) attn = P @ V   : A = bf16 P (lda = 4096), BT = V^T, bf16 out
    {
        dim3 grd(D / 256, S / 64, B);
        gemm_bf16_wmma<<<grd, 256, 0, stream>>>((const bf16_t*)Sb,
                                                (long)S * 2 * S, 2 * S,
                                                vT, (long)D * S, S,
                                                attn, (long)S * D, D,
                                                nullptr, 1.f, S, 1);
    }

    // 8) out = attn @ Wo + bo   (f32 out to d_out)
    {
        dim3 grd(D / 256, M / 64, 1);
        gemm_bf16_wmma<<<grd, 256, 0, stream>>>(attn, 0, D, WoT, 0, D,
                                                out, 0, D, bo, 1.f, D, 0);
    }
}
